// BiLSTM_CRF_63213328663037
// MI455X (gfx1250) — matrix operand order repaired and checked by reading, not compiled, not run on hardware
//
#include <hip/hip_runtime.h>
#include <hip/hip_bf16.h>

typedef __attribute__((ext_vector_type(16))) _Float16 v16h;
typedef __attribute__((ext_vector_type(8)))  float    v8f;

#define T_  256
#define B_  64
#define E_  256
#define H_  256
#define G4_ 1024   // 4*H
#define K_  10

// Map intra-32 K offset (ko) of the 16-bit A-matrix 16x32 WMMA layout to
// (half-slot j, lane-half hi).  Contiguous order (matches B packing):
//   K = j + 16*hi
__device__ __forceinline__ void a_swz_pos(int ko, int& j, int& hi) {
  j = ko & 15; hi = ko >> 4;
}

__device__ __forceinline__ float sigf(float x) { return 1.f / (1.f + __expf(-x)); }

// ---------------------------------------------------------------------------
// Convert W (1024 x 256, f32, row-major, used as B = W^T in x@W^T) into the
// CDNA5 16-bit B-matrix 32x16 fragment layout, f16.
// out idx = ((kc*64 + nt)*32 + lane)*16 + j
//   value = W[n*256 + k],  n = nt*16 + (lane&15),  k = kc*32 + j + 16*(lane>>4)
// ---------------------------------------------------------------------------
__global__ void wswz_kernel(const float* __restrict__ W, _Float16* __restrict__ out) {
  int idx  = blockIdx.x * blockDim.x + threadIdx.x;   // 262144 total
  int j    = idx & 15;
  int lane = (idx >> 4) & 31;
  int nt   = (idx >> 9) & 63;
  int kc   = idx >> 15;
  int nidx = nt * 16 + (lane & 15);
  int k    = kc * 32 + j + 16 * (lane >> 4);
  out[idx] = (_Float16)W[nidx * 256 + k];
}

__global__ void bsum_kernel(const float* __restrict__ bih, const float* __restrict__ bhh,
                            float* __restrict__ bsum) {
  int i = blockIdx.x * blockDim.x + threadIdx.x;
  if (i < G4_) bsum[i] = bih[i] + bhh[i];
}

// ---------------------------------------------------------------------------
// Embedding gather, written straight into A-fragment (16x32 f16) order, with
// rows ordered r = t*B + b so that a 16-row M-tile = 16 batch rows at fixed t.
// Also builds the per-sequence reversed copy for the backward LSTM.
// ---------------------------------------------------------------------------
__global__ void embed_kernel(const int* __restrict__ inp, const int* __restrict__ nlen,
                             const float* __restrict__ emb,
                             _Float16* __restrict__ xf, _Float16* __restrict__ xb) {
  int idx = blockIdx.x * blockDim.x + threadIdx.x;    // T*B*E = 4194304
  int e = idx & 255;
  int b = (idx >> 8) & 63;
  int t = idx >> 14;
  int nb = nlen[b];
  int tr = (t < nb) ? (nb - 1 - t) : t;
  float vf = emb[(long)inp[b * T_ + t ] * E_ + e];
  float vb = emb[(long)inp[b * T_ + tr] * E_ + e];
  int row = t * B_ + b;
  int rt  = row >> 4;
  int m   = row & 15;
  int kc = e >> 5, ko = e & 31, j, hi;
  a_swz_pos(ko, j, hi);
  int off = ((rt * 8 + kc) * 32 + (hi * 16 + m)) * 16 + j;
  xf[off] = (_Float16)vf;
  xb[off] = (_Float16)vb;
}

// ---------------------------------------------------------------------------
// Time-parallel input GEMM:  gates[(t*B+b), n] = sum_k x[row,k] * W^T[k,n]
// One wave per 16x16 C tile, K=256 -> 8 x v_wmma_f32_16x16x32_f16.
// ---------------------------------------------------------------------------
__global__ void input_gemm_kernel(const _Float16* __restrict__ xswz,
                                  const _Float16* __restrict__ wswz,
                                  _Float16* __restrict__ gates) {
  int wv   = blockIdx.x * (blockDim.x >> 5) + (threadIdx.x >> 5);
  int lane = threadIdx.x & 31;
  int rt   = wv >> 6;                 // 0..1023  (16384 rows / 16)
  int nt   = wv & 63;                 // 0..63    (1024 cols / 16)
  v8f acc = {};
#pragma unroll
  for (int kc = 0; kc < 8; ++kc) {
    v16h a = *(const v16h*)(xswz + (((rt * 8 + kc) * 32 + lane) << 4));
    v16h bf = *(const v16h*)(wswz + (((kc * 64 + nt) * 32 + lane) << 4));
    acc = __builtin_amdgcn_wmma_f32_16x16x32_f16(false, a, false, bf, (short)0, acc,
                                                 false, false);
  }
  int hi = lane >> 4, col = lane & 15;
#pragma unroll
  for (int v = 0; v < 8; ++v) {
    int m = v + 8 * hi;                        // C layout: VGPR v -> M = v + 8*hi
    gates[(long)(rt * 16 + m) * G4_ + nt * 16 + col] = (_Float16)acc[v];
  }
}

// ---------------------------------------------------------------------------
// Persistent recurrent LSTM kernel (one launch per direction).
// Grid = 4 blocks x 512 threads (16 waves). Block k owns batch rows
// [16k, 16k+16). h lives in LDS in A-fragment order; c lives in VGPRs.
// Wave w owns hidden slice [16w,16w+16) -> computes i/f/g/o tiles for it,
// so the gate nonlinearity + cell update never leaves registers.
// ---------------------------------------------------------------------------
__global__ void lstm_seq_kernel(const _Float16* __restrict__ gates_in,
                                const _Float16* __restrict__ whh_swz,
                                const float* __restrict__ bsum,
                                _Float16* __restrict__ h_out) {
  __shared__ __align__(32) _Float16 h_sh[4096];   // 16 rows x 256 k, A-frag order
  int tid  = threadIdx.x;
  int w    = tid >> 5;
  int lane = tid & 31;
  int b0   = blockIdx.x * 16;
  for (int i = tid; i < 4096; i += blockDim.x) h_sh[i] = (_Float16)0.f;
  float c_frag[8];
#pragma unroll
  for (int v = 0; v < 8; ++v) c_frag[v] = 0.f;
  __syncthreads();

  int hi = lane >> 4, col = lane & 15;
  int hcol = w * 16 + col;

  for (int t = 0; t < T_; ++t) {
    v8f ai = {}, af = {}, ag = {}, ao = {};
#pragma unroll
    for (int kc = 0; kc < 8; ++kc) {
      v16h a  = *(const v16h*)(h_sh + ((kc * 32 + lane) << 4));     // ds_load_b128 x2
      v16h bi = *(const v16h*)(whh_swz + (((kc * 64 + (w     )) * 32 + lane) << 4));
      v16h bf = *(const v16h*)(whh_swz + (((kc * 64 + (16 + w)) * 32 + lane) << 4));
      v16h bg = *(const v16h*)(whh_swz + (((kc * 64 + (32 + w)) * 32 + lane) << 4));
      v16h bo = *(const v16h*)(whh_swz + (((kc * 64 + (48 + w)) * 32 + lane) << 4));
      ai = __builtin_amdgcn_wmma_f32_16x16x32_f16(false, a, false, bi, (short)0, ai, false, false);
      af = __builtin_amdgcn_wmma_f32_16x16x32_f16(false, a, false, bf, (short)0, af, false, false);
      ag = __builtin_amdgcn_wmma_f32_16x16x32_f16(false, a, false, bg, (short)0, ag, false, false);
      ao = __builtin_amdgcn_wmma_f32_16x16x32_f16(false, a, false, bo, (short)0, ao, false, false);
    }
    __syncthreads();      // all h(t-1) reads complete before h(t) writes
#pragma unroll
    for (int v = 0; v < 8; ++v) {
      int  m = v + 8 * hi;
      long r = (long)t * B_ + b0 + m;
      float gi = (float)gates_in[r * G4_ +       hcol] + ai[v] + bsum[      hcol];
      float gf = (float)gates_in[r * G4_ + 256 + hcol] + af[v] + bsum[256 + hcol];
      float gg = (float)gates_in[r * G4_ + 512 + hcol] + ag[v] + bsum[512 + hcol];
      float go = (float)gates_in[r * G4_ + 768 + hcol] + ao[v] + bsum[768 + hcol];
      float c  = sigf(gf) * c_frag[v] + sigf(gi) * tanhf(gg);
      c_frag[v] = c;
      float h = sigf(go) * tanhf(c);
      h_out[r * H_ + hcol] = (_Float16)h;
      int kc2 = hcol >> 5, ko = hcol & 31, j2, hi2;
      a_swz_pos(ko, j2, hi2);
      h_sh[((kc2 * 32 + (hi2 * 16 + m)) << 4) + j2] = (_Float16)h;
    }
    __syncthreads();      // h(t) visible before next step's A-fragment loads
  }
}

// ---------------------------------------------------------------------------
// feats[b,t,k] = fc_b[k] + sum_h fc_w[k,h] * concat(hf, hb_unreversed)[b,t,h]
// (positions t >= n[b] are never read by Viterbi, so no masking needed here)
// ---------------------------------------------------------------------------
__global__ void fc_kernel(const _Float16* __restrict__ hf, const _Float16* __restrict__ hb,
                          const float* __restrict__ fc_w, const float* __restrict__ fc_b,
                          const int* __restrict__ nlen, float* __restrict__ feats) {
  int idx = blockIdx.x * blockDim.x + threadIdx.x;     // B*T*K = 163840
  if (idx >= B_ * T_ * K_) return;
  int k = idx % K_;
  int r = idx / K_;
  int t = r % T_;
  int b = r / T_;
  int nb = nlen[b];
  int tr = (t < nb) ? (nb - 1 - t) : t;
  float acc = fc_b[k];
  const float*    wr  = fc_w + k * 512;
  const _Float16* hfp = hf + ((long)t  * B_ + b) * H_;
  const _Float16* hbp = hb + ((long)tr * B_ + b) * H_;
  for (int h = 0; h < H_; ++h) acc += wr[h]       * (float)hfp[h];
  for (int h = 0; h < H_; ++h) acc += wr[256 + h] * (float)hbp[h];
  feats[idx] = acc;     // layout (b*T + t)*K + k
}

// ---------------------------------------------------------------------------
// Viterbi decode: one thread per batch element (K=10).  Forward pass stores
// uint8 backpointers in workspace; backtrace writes masked float tags.
// ---------------------------------------------------------------------------
__global__ void viterbi_kernel(const float* __restrict__ feats,
                               const float* __restrict__ transition,
                               const int* __restrict__ nlen,
                               unsigned char* __restrict__ bp,
                               float* __restrict__ out) {
  __shared__ float tr[K_ * K_];
  int tid = threadIdx.x;
  for (int i = tid; i < K_ * K_; i += blockDim.x)
    tr[i] = transition[(i / K_) * 12 + (i % K_)];   // trans_in = transition[:10,:10]
  __syncthreads();
  int b  = tid;
  int nb = nlen[b];
  const float* fb = feats + (long)b * T_ * K_;
  float alpha[K_], na[K_];
  for (int j = 0; j < K_; ++j) alpha[j] = fb[j];
  for (int t = 1; t < T_; ++t) {
    bool valid = t < nb;
    for (int j = 0; j < K_; ++j) {
      float best = alpha[0] + tr[j];
      int bi = 0;
      for (int i = 1; i < K_; ++i) {
        float s = alpha[i] + tr[i * K_ + j];
        if (s > best) { best = s; bi = i; }
      }
      if (valid) { na[j] = best + fb[t * K_ + j];
                   bp[((long)b * T_ + t) * K_ + j] = (unsigned char)bi; }
      else       { na[j] = alpha[j];
                   bp[((long)b * T_ + t) * K_ + j] = (unsigned char)j; }
    }
    for (int j = 0; j < K_; ++j) alpha[j] = na[j];
  }
  int tag = 0; float best = alpha[0];
  for (int j = 1; j < K_; ++j) if (alpha[j] > best) { best = alpha[j]; tag = j; }
  for (int t = T_ - 1; t >= 1; --t) {
    out[b * T_ + t] = (t < nb) ? (float)tag : 0.f;
    tag = bp[((long)b * T_ + t) * K_ + tag];
  }
  out[b * T_] = (float)tag;   // t=0 always valid (n >= 1)
}

// ---------------------------------------------------------------------------
extern "C" void kernel_launch(void* const* d_in, const int* in_sizes, int n_in,
                              void* d_out, int out_size, void* d_ws, size_t ws_size,
                              hipStream_t stream) {
  const int*   inp   = (const int*)  d_in[0];
  const int*   nlen  = (const int*)  d_in[1];
  const float* emb   = (const float*)d_in[2];
  const float* Wih_f = (const float*)d_in[3];
  const float* Whh_f = (const float*)d_in[4];
  const float* bih_f = (const float*)d_in[5];
  const float* bhh_f = (const float*)d_in[6];
  const float* Wih_b = (const float*)d_in[7];
  const float* Whh_b = (const float*)d_in[8];
  const float* bih_b = (const float*)d_in[9];
  const float* bhh_b = (const float*)d_in[10];
  const float* fc_w  = (const float*)d_in[11];
  const float* fc_b  = (const float*)d_in[12];
  const float* trans = (const float*)d_in[13];
  float* out = (float*)d_out;

  char*  base = (char*)d_ws;
  size_t off  = 0;
  auto alloc = [&](size_t bytes) -> void* {
    void* r = base + off;
    off = (off + bytes + 255) & ~(size_t)255;
    return r;
  };
  const size_t BT = (size_t)B_ * T_;
  _Float16* xswz_f  = (_Float16*)alloc(BT * E_ * 2);        // 8 MB
  _Float16* xswz_b  = (_Float16*)alloc(BT * E_ * 2);        // 8 MB
  _Float16* wih_f_s = (_Float16*)alloc((size_t)G4_ * E_ * 2);
  _Float16* whh_f_s = (_Float16*)alloc((size_t)G4_ * H_ * 2);
  _Float16* wih_b_s = (_Float16*)alloc((size_t)G4_ * E_ * 2);
  _Float16* whh_b_s = (_Float16*)alloc((size_t)G4_ * H_ * 2);
  float*    bsum_f  = (float*)   alloc(G4_ * 4);
  float*    bsum_b  = (float*)   alloc(G4_ * 4);
  _Float16* gates_f = (_Float16*)alloc(BT * G4_ * 2);       // 32 MB
  _Float16* gates_b = (_Float16*)alloc(BT * G4_ * 2);       // 32 MB
  _Float16* hf      = (_Float16*)alloc(BT * H_ * 2);        // 8 MB
  _Float16* hb      = (_Float16*)alloc(BT * H_ * 2);        // 8 MB
  float*    feats   = (float*)   alloc(BT * K_ * 4);
  unsigned char* bp = (unsigned char*)alloc(BT * K_);

  // Weight / bias prep
  wswz_kernel<<<1024, 256, 0, stream>>>(Wih_f, wih_f_s);
  wswz_kernel<<<1024, 256, 0, stream>>>(Whh_f, whh_f_s);
  wswz_kernel<<<1024, 256, 0, stream>>>(Wih_b, wih_b_s);
  wswz_kernel<<<1024, 256, 0, stream>>>(Whh_b, whh_b_s);
  bsum_kernel<<<4, 256, 0, stream>>>(bih_f, bhh_f, bsum_f);
  bsum_kernel<<<4, 256, 0, stream>>>(bih_b, bhh_b, bsum_b);

  // Embedding gather (fwd + reversed) into WMMA A-fragment layout
  embed_kernel<<<16384, 256, 0, stream>>>(inp, nlen, emb, xswz_f, xswz_b);

  // Time-parallel input GEMMs (x @ Wih^T), both directions
  input_gemm_kernel<<<8192, 256, 0, stream>>>(xswz_f, wih_f_s, gates_f);
  input_gemm_kernel<<<8192, 256, 0, stream>>>(xswz_b, wih_b_s, gates_b);

  // Persistent recurrent kernels (one launch per direction)
  lstm_seq_kernel<<<4, 512, 0, stream>>>(gates_f, whh_f_s, bsum_f, hf);
  lstm_seq_kernel<<<4, 512, 0, stream>>>(gates_b, whh_b_s, bsum_b, hb);

  // Emissions + Viterbi
  fc_kernel<<<640, 256, 0, stream>>>(hf, hb, fc_w, fc_b, nlen, feats);
  viterbi_kernel<<<1, 64, 0, stream>>>(feats, trans, nlen, bp, out);
}
